// LlamaAttention_12524124635560
// MI455X (gfx1250) — compile-verified, matrix operand order checked by reading, not run on hardware
//
#include <hip/hip_runtime.h>
#include <hip/hip_bf16.h>

// ---------------------------------------------------------------------------
// Types
// ---------------------------------------------------------------------------
typedef __bf16 bf16_t;
typedef __attribute__((ext_vector_type(16))) __bf16  v16bf;
typedef __attribute__((ext_vector_type(8)))  float   v8f;
typedef __attribute__((ext_vector_type(4)))  unsigned int u32x4;
typedef __attribute__((ext_vector_type(4)))  int          s32x4;

#define S_LEN    2048
#define D_MODEL  4096
#define HEADS    32
#define KV_HEADS 8
#define HEAD_DIM 128
#define KV_DIM   1024
#define BATCH    2
#define M_ROWS   (BATCH * S_LEN)   // 4096

// ---------------------------------------------------------------------------
// CDNA5 async global->LDS staging (ASYNCcnt-tracked), with fallback
// ---------------------------------------------------------------------------
#if defined(__AMDGCN__) && __has_builtin(__builtin_amdgcn_global_load_async_to_lds_b128)
#define HAVE_ASYNC_LDS 1
#else
#define HAVE_ASYNC_LDS 0
#endif

static __device__ __forceinline__ void stage16B(const unsigned short* g,
                                                unsigned short* l) {
#if HAVE_ASYNC_LDS
    __builtin_amdgcn_global_load_async_to_lds_b128(
        (__attribute__((address_space(1))) s32x4*)g,
        (__attribute__((address_space(3))) s32x4*)l, 0, 0);
#else
    *(u32x4*)l = *(const u32x4*)g;
#endif
}

static __device__ __forceinline__ void async_wait0() {
#if defined(__AMDGCN__)
#if __has_builtin(__builtin_amdgcn_s_wait_asynccnt)
    __builtin_amdgcn_s_wait_asynccnt(0);
#else
    asm volatile("s_wait_asynccnt 0x0" ::: "memory");
#endif
#endif
}

// ---------------------------------------------------------------------------
// bf16 helpers (raw ushort storage, f32 math)
// ---------------------------------------------------------------------------
static __device__ __forceinline__ float bf2f(unsigned short h) {
    unsigned int u = ((unsigned int)h) << 16;
    float f;
    __builtin_memcpy(&f, &u, 4);
    return f;
}
static __device__ __forceinline__ unsigned short f2bf(float f) {
    unsigned int u;
    __builtin_memcpy(&u, &f, 4);
    u = (u + 0x7FFFu + ((u >> 16) & 1u)) >> 16;  // round-to-nearest-even
    return (unsigned short)u;
}

union FragU { v16bf v; u32x4 q[2]; };

// Load a 16-element bf16 WMMA fragment as two contiguous 16B runs.
// Per CDNA5 ISA 16-bit A/B layout: lane holds (M or N)=lane&15, K runs
// [kbase, kbase+8) and [kbase+16, kbase+24) with kbase = (lane>>4)*8.
static __device__ __forceinline__ v16bf ld_frag(const unsigned short* p0,
                                                const unsigned short* p1) {
    FragU f;
    f.q[0] = *(const u32x4*)p0;
    f.q[1] = *(const u32x4*)p1;
    return f.v;
}

// ---------------------------------------------------------------------------
// Kernel 1: f32 -> bf16 conversion (4 elements / thread)
// ---------------------------------------------------------------------------
__global__ void cvt_f32_bf16_kernel(const float* __restrict__ in,
                                    unsigned short* __restrict__ out,
                                    long long n4) {
    long long i = (long long)blockIdx.x * blockDim.x + threadIdx.x;
    if (i < n4) {
        const float4 v = ((const float4*)in)[i];
        ushort4 o;
        o.x = f2bf(v.x); o.y = f2bf(v.y); o.z = f2bf(v.z); o.w = f2bf(v.w);
        ((ushort4*)out)[i] = o;
    }
}

// ---------------------------------------------------------------------------
// Kernel 2: bf16 GEMM, f32 accumulate.  C[M,N] = A[M,K] * B[K,N]
// Block tile 128x128x64, 256 threads = 8 waves, each wave 32x64.
// 16 WMMA per wave per LDS tile; A tile staged via async-LDS path.
// OUT_BF16: store bf16 (activations) or f32 (final output).
// ---------------------------------------------------------------------------
template <int OUT_BF16>
__global__ __launch_bounds__(256) void gemm_bf16_kernel(
    const unsigned short* __restrict__ A, int lda,
    const unsigned short* __restrict__ B, int ldb,
    void* __restrict__ Cout, int ldc, int K) {
    __shared__ __align__(16) unsigned short As[128][64];   // [m][k]   16 KB
    __shared__ __align__(16) unsigned short Bs[128][64];   // [n][k]   16 KB

    const int tid  = threadIdx.x;
    const int lane = tid & 31;
    const int wave = tid >> 5;
    const int wm   = wave >> 1;           // 0..3
    const int wn   = wave & 1;            // 0..1
    const int bm   = blockIdx.y * 128;
    const int bn   = blockIdx.x * 128;

    const int m_l = lane & 15;
    const int kb8 = (lane >> 4) * 8;

    const int arow = tid >> 1;            // 0..127
    const int acol = (tid & 1) * 32;      // 0,32
    const int bkrow  = tid >> 2;          // 0..63
    const int bchunk = (tid & 3) * 32;    // 0..96

    v8f acc[2][4];
    #pragma unroll
    for (int sm = 0; sm < 2; ++sm)
        #pragma unroll
        for (int sn = 0; sn < 4; ++sn)
            #pragma unroll
            for (int j = 0; j < 8; ++j) acc[sm][sn][j] = 0.0f;

    for (int kt = 0; kt < K; kt += 64) {
        // --- stage A tile (row-major, no transform): async global->LDS ---
        const unsigned short* ga = A + (size_t)(bm + arow) * lda + kt + acol;
        #pragma unroll
        for (int c = 0; c < 4; ++c)
            stage16B(ga + c * 8, &As[arow][acol + c * 8]);

        // --- stage B tile, transposed to [n][k] (VGPR round-trip) ---
        const unsigned short* gb = B + (size_t)(kt + bkrow) * ldb + bn + bchunk;
        __align__(16) unsigned short tmp[32];
        #pragma unroll
        for (int c = 0; c < 4; ++c)
            *(u32x4*)&tmp[c * 8] = *(const u32x4*)(gb + c * 8);
        #pragma unroll
        for (int i = 0; i < 32; ++i) Bs[bchunk + i][bkrow] = tmp[i];

        async_wait0();
        __syncthreads();

        #pragma unroll
        for (int kk = 0; kk < 2; ++kk) {
            const int kb = kk * 32 + kb8;
            v16bf af[2], bfv[4];
            #pragma unroll
            for (int sm = 0; sm < 2; ++sm) {
                const unsigned short* p = &As[wm * 32 + sm * 16 + m_l][kb];
                af[sm] = ld_frag(p, p + 16);
            }
            #pragma unroll
            for (int sn = 0; sn < 4; ++sn) {
                const unsigned short* p = &Bs[wn * 64 + sn * 16 + m_l][kb];
                bfv[sn] = ld_frag(p, p + 16);
            }
            #pragma unroll
            for (int sm = 0; sm < 2; ++sm)
                #pragma unroll
                for (int sn = 0; sn < 4; ++sn)
                    acc[sm][sn] = __builtin_amdgcn_wmma_f32_16x16x32_bf16(
                        false, af[sm], false, bfv[sn], (short)0, acc[sm][sn],
                        false, false);
        }
        __syncthreads();
    }

    // --- epilogue: C layout lane = col n, VGPR r = row (+8 for hi half) ---
    const int hi = lane >> 4;
    #pragma unroll
    for (int sm = 0; sm < 2; ++sm)
        #pragma unroll
        for (int sn = 0; sn < 4; ++sn)
            #pragma unroll
            for (int r = 0; r < 8; ++r) {
                int row = bm + wm * 32 + sm * 16 + r + 8 * hi;
                int col = bn + wn * 64 + sn * 16 + m_l;
                float v = acc[sm][sn][r];
                if (OUT_BF16)
                    ((unsigned short*)Cout)[(size_t)row * ldc + col] = f2bf(v);
                else
                    ((float*)Cout)[(size_t)row * ldc + col] = v;
            }
}

// ---------------------------------------------------------------------------
// Kernel 3: RoPE in place on bf16 activations (one thread per rotation pair).
// scale folds the attention 1/sqrt(hd) into Q.
// ---------------------------------------------------------------------------
__global__ void rope_kernel(unsigned short* __restrict__ X, int nheads,
                            int rowlen, long long total, float scale) {
    long long i = (long long)blockIdx.x * blockDim.x + threadIdx.x;
    if (i >= total) return;
    int d = (int)(i & 63);
    int h = (int)((i >> 6) % nheads);
    long long row = i / (64LL * nheads);
    int s = (int)(row % S_LEN);

    size_t off = (size_t)row * rowlen + (size_t)h * HEAD_DIM + d;
    float inv = __powf(10000.0f, -(float)d * (1.0f / 64.0f));
    float sn, cs;
    __sincosf((float)s * inv, &sn, &cs);
    float x0 = bf2f(X[off]);
    float x1 = bf2f(X[off + 64]);
    X[off]      = f2bf((x0 * cs - x1 * sn) * scale);
    X[off + 64] = f2bf((x1 * cs + x0 * sn) * scale);
}

// ---------------------------------------------------------------------------
// Kernel 4: causal GQA flash attention.
// grid (S/128, HEADS, BATCH); 256 threads = 8 waves; wave owns 16 q rows.
// K tiles of 64 keys staged via async-LDS; V staged transposed; P through
// per-wave LDS buffer to convert WMMA C layout -> A layout.
// ---------------------------------------------------------------------------
__global__ __launch_bounds__(256) void flash_attn_kernel(
    const unsigned short* __restrict__ Q,    // [B*S, 4096] (rope'd, pre-scaled)
    const unsigned short* __restrict__ Kx,   // [B*S, 1024] (rope'd)
    const unsigned short* __restrict__ Vx,   // [B*S, 1024]
    unsigned short* __restrict__ Ctx) {      // [B*S, 4096]
    __shared__ __align__(16) unsigned short Ks[64][128];    // [k][d]
    __shared__ __align__(16) unsigned short Vt[128][64];    // [d][k]
    __shared__ __align__(16) unsigned short Pb[8][16][72];  // per-wave P

    const int tid  = threadIdx.x;
    const int lane = tid & 31;
    const int wave = tid >> 5;
    const int qt   = blockIdx.x;
    const int h    = blockIdx.y;
    const int b    = blockIdx.z;
    const int kvh  = h >> 2;                // GQA: 4 q-heads per kv-head

    const int m_l = lane & 15;
    const int hi  = lane >> 4;
    const int kb8 = hi * 8;

    const size_t qrow0 = (size_t)b * S_LEN + (size_t)qt * 128 + wave * 16;
    const unsigned short* qbase = Q + qrow0 * D_MODEL + h * HEAD_DIM;

    // Q fragments, direct from global (A layout == row-major contiguous runs)
    v16bf aq[4];
    #pragma unroll
    for (int dc = 0; dc < 4; ++dc) {
        const unsigned short* p = qbase + (size_t)m_l * D_MODEL + dc * 32 + kb8;
        aq[dc] = ld_frag(p, p + 16);
    }

    v8f o[8];
    #pragma unroll
    for (int dt = 0; dt < 8; ++dt)
        #pragma unroll
        for (int j = 0; j < 8; ++j) o[dt][j] = 0.0f;
    float mst[8], lst[8];
    #pragma unroll
    for (int r = 0; r < 8; ++r) { mst[r] = -1e30f; lst[r] = 0.0f; }

    const int    ktiles = (qt + 1) * 2;      // uniform per block (causal bound)
    const size_t kvrow0 = (size_t)b * S_LEN;
    const int    ldrow  = tid >> 2;          // 0..63
    const int    ldcol  = (tid & 3) * 32;    // 0..96

    for (int t = 0; t < ktiles; ++t) {
        __syncthreads();
        // stage K tile [64][128]: async global->LDS (no transform)
        const unsigned short* gk =
            Kx + (kvrow0 + (size_t)t * 64 + ldrow) * KV_DIM + kvh * HEAD_DIM + ldcol;
        #pragma unroll
        for (int c = 0; c < 4; ++c)
            stage16B(gk + c * 8, &Ks[ldrow][ldcol + c * 8]);
        // stage V tile transposed -> Vt[d][k]
        const unsigned short* gv =
            Vx + (kvrow0 + (size_t)t * 64 + ldrow) * KV_DIM + kvh * HEAD_DIM + ldcol;
        __align__(16) unsigned short vtmp[32];
        #pragma unroll
        for (int c = 0; c < 4; ++c)
            *(u32x4*)&vtmp[c * 8] = *(const u32x4*)(gv + c * 8);
        #pragma unroll
        for (int i = 0; i < 32; ++i) Vt[ldcol + i][ldrow] = vtmp[i];

        async_wait0();
        __syncthreads();

        // ---- S = Q K^T : 4 n-tiles x 4 k(d)-chunks ----
        v8f st[4];
        #pragma unroll
        for (int nt = 0; nt < 4; ++nt) {
            #pragma unroll
            for (int j = 0; j < 8; ++j) st[nt][j] = 0.0f;
            const unsigned short* pk = &Ks[nt * 16 + m_l][kb8];
            #pragma unroll
            for (int dc = 0; dc < 4; ++dc) {
                v16bf bk = ld_frag(pk + dc * 32, pk + dc * 32 + 16);
                st[nt] = __builtin_amdgcn_wmma_f32_16x16x32_bf16(
                    false, aq[dc], false, bk, (short)0, st[nt], false, false);
            }
        }

        // ---- causal mask + online softmax ----
        const int k0        = t * 64;
        const int qpos_base = qt * 128 + wave * 16;
        #pragma unroll
        for (int r = 0; r < 8; ++r) {
            const int qp = qpos_base + r + 8 * hi;
            float tm = -1e30f;
            #pragma unroll
            for (int nt = 0; nt < 4; ++nt) {
                int   kp = k0 + nt * 16 + m_l;
                float sv = st[nt][r];
                if (kp > qp) sv = -1e30f;
                st[nt][r] = sv;
                tm = fmaxf(tm, sv);
            }
            tm = fmaxf(tm, __shfl_xor(tm, 1, 32));
            tm = fmaxf(tm, __shfl_xor(tm, 2, 32));
            tm = fmaxf(tm, __shfl_xor(tm, 4, 32));
            tm = fmaxf(tm, __shfl_xor(tm, 8, 32));

            float mn    = fmaxf(mst[r], tm);
            float alpha = __expf(mst[r] - mn);
            mst[r]      = mn;

            float rs = 0.0f;
            #pragma unroll
            for (int nt = 0; nt < 4; ++nt) {
                float p = __expf(st[nt][r] - mn);
                st[nt][r] = p;
                rs += p;
            }
            rs += __shfl_xor(rs, 1, 32);
            rs += __shfl_xor(rs, 2, 32);
            rs += __shfl_xor(rs, 4, 32);
            rs += __shfl_xor(rs, 8, 32);
            lst[r] = lst[r] * alpha + rs;

            #pragma unroll
            for (int dt = 0; dt < 8; ++dt) o[dt][r] *= alpha;

            const int m = r + 8 * hi;
            #pragma unroll
            for (int nt = 0; nt < 4; ++nt)
                Pb[wave][m][nt * 16 + m_l] = f2bf(st[nt][r]);
        }

        // ---- O += P V : 2 k-chunks x 8 d-tiles ----
        #pragma unroll
        for (int kc = 0; kc < 2; ++kc) {
            const unsigned short* pp = &Pb[wave][m_l][kc * 32 + kb8];
            v16bf ap = ld_frag(pp, pp + 16);
            #pragma unroll
            for (int dt = 0; dt < 8; ++dt) {
                const unsigned short* pv = &Vt[dt * 16 + m_l][kc * 32 + kb8];
                v16bf bv = ld_frag(pv, pv + 16);
                o[dt] = __builtin_amdgcn_wmma_f32_16x16x32_bf16(
                    false, ap, false, bv, (short)0, o[dt], false, false);
            }
        }
    }

    // ---- finalize: O / l -> context (bf16) ----
    #pragma unroll
    for (int r = 0; r < 8; ++r) {
        float  inv = 1.0f / lst[r];
        size_t row = qrow0 + r + 8 * hi;
        #pragma unroll
        for (int dt = 0; dt < 8; ++dt)
            Ctx[row * D_MODEL + h * HEAD_DIM + dt * 16 + m_l] =
                f2bf(o[dt][r] * inv);
    }
}

// ---------------------------------------------------------------------------
// Host orchestration
// ---------------------------------------------------------------------------
extern "C" void kernel_launch(void* const* d_in, const int* in_sizes, int n_in,
                              void* d_out, int out_size, void* d_ws,
                              size_t ws_size, hipStream_t stream) {
    (void)in_sizes; (void)n_in; (void)out_size; (void)ws_size;
    const float* x  = (const float*)d_in[0];
    const float* wq = (const float*)d_in[1];
    const float* wk = (const float*)d_in[2];
    const float* wv = (const float*)d_in[3];
    const float* wo = (const float*)d_in[4];
    // d_in[5] = attention_mask: exactly causal, handled analytically.

    char*  ws  = (char*)d_ws;
    size_t off = 0;
    auto alloc = [&](size_t bytes) -> unsigned short* {
        unsigned short* p = (unsigned short*)(ws + off);
        off += (bytes + 255) & ~(size_t)255;
        return p;
    };
    const size_t MD  = (size_t)M_ROWS * D_MODEL;
    const size_t MKV = (size_t)M_ROWS * KV_DIM;

    unsigned short* Xb   = alloc(MD * 2);
    unsigned short* Wqb  = alloc((size_t)D_MODEL * D_MODEL * 2);
    unsigned short* Wkb  = alloc((size_t)D_MODEL * KV_DIM * 2);
    unsigned short* Wvb  = alloc((size_t)D_MODEL * KV_DIM * 2);
    unsigned short* Wob  = alloc((size_t)D_MODEL * D_MODEL * 2);
    unsigned short* Qb   = alloc(MD * 2);
    unsigned short* Kbuf = alloc(MKV * 2);
    unsigned short* Vbuf = alloc(MKV * 2);
    unsigned short* Ctx  = alloc(MD * 2);

    auto cvt = [&](const float* in, unsigned short* outp, long long n) {
        long long n4 = n / 4;
        cvt_f32_bf16_kernel<<<(unsigned)((n4 + 255) / 256), 256, 0, stream>>>(
            in, outp, n4);
    };
    cvt(x,  Xb,  (long long)MD);
    cvt(wq, Wqb, (long long)D_MODEL * D_MODEL);
    cvt(wk, Wkb, (long long)D_MODEL * KV_DIM);
    cvt(wv, Wvb, (long long)D_MODEL * KV_DIM);
    cvt(wo, Wob, (long long)D_MODEL * D_MODEL);

    // QKV projections (bf16 out)
    dim3 gq(D_MODEL / 128, M_ROWS / 128);   // (32, 32)
    dim3 gkv(KV_DIM / 128, M_ROWS / 128);   // (8, 32)
    gemm_bf16_kernel<1><<<gq, 256, 0, stream>>>(Xb, D_MODEL, Wqb, D_MODEL, Qb,
                                                D_MODEL, D_MODEL);
    gemm_bf16_kernel<1><<<gkv, 256, 0, stream>>>(Xb, D_MODEL, Wkb, KV_DIM, Kbuf,
                                                 KV_DIM, D_MODEL);
    gemm_bf16_kernel<1><<<gkv, 256, 0, stream>>>(Xb, D_MODEL, Wvb, KV_DIM, Vbuf,
                                                 KV_DIM, D_MODEL);

    // RoPE (fold 1/sqrt(128) into Q)
    {
        long long tq = (long long)M_ROWS * HEADS * 64;
        rope_kernel<<<(unsigned)((tq + 255) / 256), 256, 0, stream>>>(
            Qb, HEADS, D_MODEL, tq, 0.08838834764831845f);
        long long tk = (long long)M_ROWS * KV_HEADS * 64;
        rope_kernel<<<(unsigned)((tk + 255) / 256), 256, 0, stream>>>(
            Kbuf, KV_HEADS, KV_DIM, tk, 1.0f);
    }

    // Flash attention
    flash_attn_kernel<<<dim3(S_LEN / 128, HEADS, BATCH), 256, 0, stream>>>(
        Qb, Kbuf, Vbuf, Ctx);

    // Output projection (f32 out)
    gemm_bf16_kernel<0><<<gq, 256, 0, stream>>>(Ctx, D_MODEL, Wob, D_MODEL,
                                                d_out, D_MODEL, D_MODEL);
}
